// AtomCenteredTensorMomentDescriptor_48833778155683
// MI455X (gfx1250) — matrix operand order repaired and checked by reading, not compiled, not run on hardware
//
#include <hip/hip_runtime.h>

// ---------------- CDNA5 WMMA types ----------------
typedef __attribute__((ext_vector_type(16))) __bf16 v16bf;
typedef __attribute__((ext_vector_type(8)))  __bf16 v8bf;
typedef __attribute__((ext_vector_type(8)))  float  v8f;

// ---------------- Problem constants ----------------
#define L_MAX     4
#define NUM_LM    25
#define FEAT      64
#define NUM_RBF   16
#define NUM_ATOMS 2000
#define NUM_EDGES 40000
#define N_PATHS   42
#define R_CUT     5.0f
#define MAX_ENT   8192

// ws layout (bytes)
#define OFF_NNZ      0
#define OFF_ENTRIES  256
#define OFF_WT       (131072)                 // 6 sets * 5*64*64 bf16 = 245760 B
#define OFF_AATOM    (393216)                 // 2000*64 f32 = 512000 B
#define OFF_BUF      (1048576)
#define BUF_ELEMS    ((size_t)NUM_LM * NUM_EDGES * FEAT)   // 64e6 elements
#define BUF_BYTES    (BUF_ELEMS * 2)                       // bf16 buffers: 128 MB

struct CGEntry { unsigned char iu, iv, io, p; float val; };

__device__ __constant__ unsigned char c_DEG[NUM_LM] =
  {0, 1,1,1, 2,2,2,2,2, 3,3,3,3,3,3,3, 4,4,4,4,4,4,4,4,4};

// float -> bf16 (round-to-nearest-even) via bit ops
static __device__ inline __bf16 f2bf(float f) {
  unsigned u = __builtin_bit_cast(unsigned, f);
  u += 0x7FFFu + ((u >> 16) & 1u);
  unsigned short h = (unsigned short)(u >> 16);
  return __builtin_bit_cast(__bf16, h);
}
// bf16 -> float via bit shift
static __device__ inline float bf2f(__bf16 b) {
  unsigned short h = __builtin_bit_cast(unsigned short, b);
  unsigned u = ((unsigned)h) << 16;
  return __builtin_bit_cast(float, u);
}

// ============================================================
// Kernel 1: compute CG nonzero table (single thread, doubles).
// Enumerates paths in sorted (l1,l2,l3) order to match PATH_KEYS.
// ============================================================
static __device__ double dfact(int n) { double r = 1.0; for (int i = 2; i <= n; ++i) r *= (double)i; return r; }

static __device__ double cg_complex(int l1, int m1, int l2, int m2, int l3, int m3) {
  int ad = l1 - l2; if (ad < 0) ad = -ad;
  if (m3 != m1 + m2 || l3 < ad || l3 > l1 + l2) return 0.0;
  double pre = (double)(2 * l3 + 1) * dfact(l1 + l2 - l3) * dfact(l1 - l2 + l3) *
               dfact(-l1 + l2 + l3) / dfact(l1 + l2 + l3 + 1);
  pre = sqrt(pre * dfact(l1 + m1) * dfact(l1 - m1) * dfact(l2 + m2) * dfact(l2 - m2) *
             dfact(l3 + m3) * dfact(l3 - m3));
  double s = 0.0;
  for (int k = 0; k <= l1 + l2 - l3; ++k) {
    int d0 = k, d1 = l1 + l2 - l3 - k, d2 = l1 - m1 - k, d3 = l2 + m2 - k;
    int d4 = l3 - l2 + m1 + k, d5 = l3 - l1 - m2 + k;
    if (d0 < 0 || d1 < 0 || d2 < 0 || d3 < 0 || d4 < 0 || d5 < 0) continue;
    double den = dfact(d0) * dfact(d1) * dfact(d2) * dfact(d3) * dfact(d4) * dfact(d5);
    s += ((k & 1) ? -1.0 : 1.0) / den;
  }
  return pre * s;
}

// nonzeros of row a of U_real(l): up to 2 (col, re, im) entries
static __device__ int urow(int l, int a, int* col, double* re, double* im) {
  const double s = 0.70710678118654752440;
  int m = a - l;
  if (m == 0) { col[0] = l; re[0] = 1.0; im[0] = 0.0; return 1; }
  if (m > 0) {
    col[0] =  m + l; re[0] = ((m & 1) ? -s : s); im[0] = 0.0;
    col[1] = -m + l; re[1] = s;                  im[1] = 0.0;
    return 2;
  }
  col[0] =  m + l; re[0] = 0.0; im[0] = s;
  col[1] = -m + l; re[1] = 0.0; im[1] = -((((-m) & 1) ? -1.0 : 1.0)) * s;
  return 2;
}

__global__ void init_cg_tables(CGEntry* __restrict__ entries, int* __restrict__ nnz_out) {
  if (threadIdx.x != 0 || blockIdx.x != 0) return;
  int cnt = 0, p = 0;
  for (int l1 = 0; l1 <= L_MAX; ++l1)
    for (int l2 = 0; l2 <= L_MAX; ++l2) {
      int lo = (l1 > l2) ? l1 - l2 : l2 - l1;
      int hi = (l1 + l2 < L_MAX) ? l1 + l2 : L_MAX;
      for (int l3 = lo; l3 <= hi; ++l3) {
        if ((l1 + l2 + l3) & 1) continue;
        for (int a = 0; a <= 2 * l1; ++a)
          for (int b = 0; b <= 2 * l2; ++b)
            for (int c = 0; c <= 2 * l3; ++c) {
              int uc[2], vc[2], wc[2];
              double ure[2], uim[2], vre[2], vim[2], wre[2], wim[2];
              int nu = urow(l1, a, uc, ure, uim);
              int nv = urow(l2, b, vc, vre, vim);
              int nw = urow(l3, c, wc, wre, wim);
              double accr = 0.0;
              for (int iu = 0; iu < nu; ++iu)
                for (int iv = 0; iv < nv; ++iv)
                  for (int iw = 0; iw < nw; ++iw) {
                    double cg = cg_complex(l1, uc[iu] - l1, l2, vc[iv] - l2, l3, wc[iw] - l3);
                    if (cg == 0.0) continue;
                    double pr = ure[iu] * vre[iv] - uim[iu] * vim[iv];
                    double pi = ure[iu] * vim[iv] + uim[iu] * vre[iv];
                    accr += (pr * wre[iw] + pi * wim[iw]) * cg;   // * conj(w)
                  }
              if (fabs(accr) > 1e-10 && cnt < MAX_ENT) {
                CGEntry en;
                en.iu = (unsigned char)(l1 * l1 + a);
                en.iv = (unsigned char)(l2 * l2 + b);
                en.io = (unsigned char)(l3 * l3 + c);
                en.p  = (unsigned char)p;
                en.val = (float)accr;
                entries[cnt++] = en;
              }
            }
        ++p;
      }
    }
  *nnz_out = cnt;
}

// ============================================================
// Kernel 2: convert + transpose deg weights f32[l][k][g] -> bf16[l][g][k]
// ============================================================
__global__ void wconv_kernel(const float* __restrict__ W, __bf16* __restrict__ Wt) {
  int i = blockIdx.x * blockDim.x + threadIdx.x;
  if (i >= 5 * 64 * 64) return;
  int l = i >> 12, k = (i >> 6) & 63, g = i & 63;
  Wt[l * 4096 + g * 64 + k] = f2bf(W[i]);
}

// ============================================================
// Kernel 3: radial features -> Y[lm*E*64 + e*64 + f]  (bf16)
// ============================================================
__global__ void radial_kernel(const float* __restrict__ disp,
                              const int* __restrict__ nbr,
                              const int* __restrict__ Z,
                              const float* __restrict__ W_rad,
                              __bf16* __restrict__ Y) {
  int tid = threadIdx.x;
  int f = tid & 63;
  int e = blockIdx.x * 4 + (tid >> 6);
  float dx = disp[e * 3 + 0], dy = disp[e * 3 + 1], dz = disp[e * 3 + 2];
  float r = sqrtf(dx * dx + dy * dy + dz * dz + 1e-12f);
  float inv = 1.0f / r;
  float x = dx * inv, y = dy * inv, z = dz * inv;
  float x2 = x * x, y2 = y * y, z2 = z * z;
  int Zj = Z[nbr[e * 2 + 1]];

  const float gamma = (NUM_RBF / R_CUT) * (NUM_RBF / R_CUT);
  float fc = (r < R_CUT) ? 0.5f * (__cosf(3.14159265358979323846f * r / R_CUT) + 1.0f) : 0.0f;
  float rad = 0.0f;
  const float* Wr = W_rad + (size_t)Zj * NUM_RBF * FEAT + f;
#pragma unroll
  for (int k = 0; k < NUM_RBF; ++k) {
    float mu = R_CUT * (float)k / (float)(NUM_RBF - 1);
    float d = r - mu;
    rad += __expf(-gamma * d * d) * fc * Wr[k * FEAT];
  }

  float sh[NUM_LM];
  sh[0] = 0.28209479177387814f;
  sh[1] = 0.4886025119029199f * y;
  sh[2] = 0.4886025119029199f * z;
  sh[3] = 0.4886025119029199f * x;
  sh[4] = 1.0925484305920792f * x * y;
  sh[5] = 1.0925484305920792f * y * z;
  sh[6] = 0.31539156525252005f * (3.0f * z2 - 1.0f);
  sh[7] = 1.0925484305920792f * x * z;
  sh[8] = 0.5462742152960396f * (x2 - y2);
  sh[9]  = 0.5900435899266435f * y * (3.0f * x2 - y2);
  sh[10] = 2.890611442640554f * x * y * z;
  sh[11] = 0.4570457994644658f * y * (5.0f * z2 - 1.0f);
  sh[12] = 0.3731763325901154f * z * (5.0f * z2 - 3.0f);
  sh[13] = 0.4570457994644658f * x * (5.0f * z2 - 1.0f);
  sh[14] = 1.445305721320277f * z * (x2 - y2);
  sh[15] = 0.5900435899266435f * x * (x2 - 3.0f * y2);
  sh[16] = 2.5033429417967046f * x * y * (x2 - y2);
  sh[17] = 1.7701307697799304f * y * z * (3.0f * x2 - y2);
  sh[18] = 0.9461746957575601f * x * y * (7.0f * z2 - 1.0f);
  sh[19] = 0.6690465435572892f * y * z * (7.0f * z2 - 3.0f);
  sh[20] = 0.10578554691520431f * (35.0f * z2 * z2 - 30.0f * z2 + 3.0f);
  sh[21] = 0.6690465435572892f * x * z * (7.0f * z2 - 3.0f);
  sh[22] = 0.47308734787878004f * (x2 - y2) * (7.0f * z2 - 1.0f);
  sh[23] = 1.7701307697799304f * x * z * (x2 - 3.0f * y2);
  sh[24] = 0.6258357354491761f * (x2 * x2 - 6.0f * x2 * y2 + y2 * y2);

  size_t base = (size_t)e * FEAT + f;
#pragma unroll
  for (int lm = 0; lm < NUM_LM; ++lm)
    Y[(size_t)lm * NUM_EDGES * FEAT + base] = f2bf(sh[lm] * rad);
}

// ============================================================
// Kernel 4: deg_dense via bf16 WMMA.  One wave computes a 32x64 tile
// (two 16-row M-tiles share each B fragment -> 16 v_wmma per wave):
//   O[r, :] = sum_k X[r, k] * W[l(r)][k, :]   (f32 accumulate, bf16 I/O)
// ============================================================
__global__ void __launch_bounds__(256)
deg_dense_wmma(const __bf16* __restrict__ X,
               const __bf16* __restrict__ Wt,   // [5][g=64][k=64] bf16
               __bf16* __restrict__ O, int total_tiles) {
  int wave = blockIdx.x * (blockDim.x >> 5) + (threadIdx.x >> 5);
  if (wave >= total_tiles) return;       // wave-uniform: EXEC stays all-ones
  int lane = threadIdx.x & 31;
  int m  = lane & 15;
  int hi = lane >> 4;

  int r0 = wave << 5;                    // 32 rows per wave
  int lm = r0 / NUM_EDGES;
  int l = 0;
  while ((l + 1) * (l + 1) <= lm) ++l;
  const __bf16* Wl = Wt + (size_t)l * 4096;

  const __bf16* Xrow0 = X + (size_t)(r0 + m) * FEAT;
  const __bf16* Xrow1 = Xrow0 + 16 * FEAT;
  __builtin_prefetch(Xrow0 + 32 * FEAT, 0, 0);   // next tile -> global_prefetch_b8

  v8f acc[2][4];
#pragma unroll
  for (int mt = 0; mt < 2; ++mt)
#pragma unroll
    for (int nt = 0; nt < 4; ++nt)
      acc[mt][nt] = (v8f){0.f, 0.f, 0.f, 0.f, 0.f, 0.f, 0.f, 0.f};

#pragma unroll
  for (int kk = 0; kk < 2; ++kk) {
    // A fragments: 16x32 bf16, contiguous k order (same as B)
    //   lanes<16: K = kk*32 + {0..15}; lanes>=16: +16
    // -> two contiguous 8-element (16B) runs: direct b128 loads, no cvt.
    v16bf a0, a1;
    {
      const __bf16* xp0 = Xrow0 + kk * 32 + hi * 16;
      const __bf16* xp1 = Xrow1 + kk * 32 + hi * 16;
      v8bf lo0 = *(const v8bf*)(xp0);
      v8bf hi0 = *(const v8bf*)(xp0 + 8);
      v8bf lo1 = *(const v8bf*)(xp1);
      v8bf hi1 = *(const v8bf*)(xp1 + 8);
#pragma unroll
      for (int t = 0; t < 8; ++t) { a0[t] = lo0[t]; a0[8 + t] = hi0[t]; }
#pragma unroll
      for (int t = 0; t < 8; ++t) { a1[t] = lo1[t]; a1[8 + t] = hi1[t]; }
    }

#pragma unroll
    for (int nt = 0; nt < 4; ++nt) {
      // B fragment: 32x16; lane = column n; K contiguous in transposed weights
      int n = nt * 16 + m;
      const __bf16* wp = Wl + (size_t)n * 64 + kk * 32 + hi * 16;
      v16bf b = *(const v16bf*)(wp);   // 32B contiguous, 32B aligned
      acc[0][nt] = __builtin_amdgcn_wmma_f32_16x16x32_bf16(
          false, a0, false, b, (short)0, acc[0][nt], false, false);
      acc[1][nt] = __builtin_amdgcn_wmma_f32_16x16x32_bf16(
          false, a1, false, b, (short)0, acc[1][nt], false, false);
    }
  }

  // C/D layout: VGPR g, lanes<16 -> M=g, lanes>=16 -> M=g+8; N = lane&15
#pragma unroll
  for (int mt = 0; mt < 2; ++mt)
#pragma unroll
    for (int nt = 0; nt < 4; ++nt) {
      int col = nt * 16 + m;
#pragma unroll
      for (int g = 0; g < 8; ++g) {
        int row = r0 + mt * 16 + g + hi * 8;
        O[(size_t)row * FEAT + col] = f2bf(acc[mt][nt][g]);
      }
    }
}

// ============================================================
// Kernel 5: sparse CG contraction (bf16 in/out, f32 LDS accumulate).
// 128 threads = 2 edges x 64 features; 3*25*128*4 = 38.4 KB LDS.
// ============================================================
__global__ void __launch_bounds__(128)
cg_kernel(const __bf16* __restrict__ U, const __bf16* __restrict__ V,
          const float* __restrict__ wp,                 // [42][64]
          __bf16* __restrict__ O,
          const CGEntry* __restrict__ entries, const int* __restrict__ nnz_ptr) {
  __shared__ float su[NUM_LM * 128];
  __shared__ float sv[NUM_LM * 128];
  __shared__ float so[NUM_LM * 128];
  int tid = threadIdx.x;
  int f = tid & 63;
  int e = blockIdx.x * 2 + (tid >> 6);
  size_t base = (size_t)e * FEAT + f;

#pragma unroll
  for (int lm = 0; lm < NUM_LM; ++lm) {
    su[lm * 128 + tid] = bf2f(U[(size_t)lm * NUM_EDGES * FEAT + base]);
    sv[lm * 128 + tid] = bf2f(V[(size_t)lm * NUM_EDGES * FEAT + base]);
    so[lm * 128 + tid] = 0.0f;
  }

  int nnz = *nnz_ptr;
  for (int n = 0; n < nnz; ++n) {
    CGEntry ent = entries[n];
    float c = ent.val * wp[(int)ent.p * FEAT + f];
    so[(int)ent.io * 128 + tid] += c * su[(int)ent.iu * 128 + tid] * sv[(int)ent.iv * 128 + tid];
  }

#pragma unroll
  for (int lm = 0; lm < NUM_LM; ++lm)
    O[(size_t)lm * NUM_EDGES * FEAT + base] = f2bf(so[lm * 128 + tid]);
}

// ============================================================
// Kernel 6: per-atom embedding  A[n][f] = emb[Z[n]] @ W_emb_t  (f32)
// ============================================================
__global__ void atom_embed(const int* __restrict__ Z, const float* __restrict__ emb,
                           const float* __restrict__ Wemb, float* __restrict__ A) {
  int i = blockIdx.x * blockDim.x + threadIdx.x;
  if (i >= NUM_ATOMS * FEAT) return;
  int n = i >> 6, f = i & 63;
  const float* er = emb + (size_t)Z[n] * FEAT;
  float s = 0.0f;
#pragma unroll 8
  for (int k = 0; k < FEAT; ++k) s += er[k] * Wemb[k * FEAT + f];
  A[i] = s;
}

// Kernel 7: initialize output (self term at lm==0, zeros elsewhere)
__global__ void out_init(const float* __restrict__ A, float* __restrict__ out) {
  int i = blockIdx.x * blockDim.x + threadIdx.x;
  if (i >= NUM_ATOMS * NUM_LM * FEAT) return;
  int f = i & 63;
  int lm = (i >> 6) % NUM_LM;
  int n = i / (NUM_LM * FEAT);
  out[i] = (lm == 0) ? A[n * FEAT + f] : 0.0f;
}

// Kernel 8: scale + segment-sum scatter (f32 atomics)
__global__ void scatter_kernel(const __bf16* __restrict__ Y, const float* __restrict__ A,
                               const float* __restrict__ wt,   // [5][64]
                               const int* __restrict__ nbr, float* __restrict__ out) {
  int tid = threadIdx.x;
  int f = tid & 63;
  int e = blockIdx.x * 4 + (tid >> 6);
  int i = nbr[e * 2 + 0];
  float a = A[i * FEAT + f];
  float* op = out + (size_t)i * NUM_LM * FEAT + f;
  size_t base = (size_t)e * FEAT + f;
#pragma unroll
  for (int lm = 0; lm < NUM_LM; ++lm) {
    float w = wt[(int)c_DEG[lm] * FEAT + f];
    float val = bf2f(Y[(size_t)lm * NUM_EDGES * FEAT + base]) * a * w;
    atomicAdd(op + lm * FEAT, val);
  }
}

// ============================================================
extern "C" void kernel_launch(void* const* d_in, const int* in_sizes, int n_in,
                              void* d_out, int out_size, void* d_ws, size_t ws_size,
                              hipStream_t stream) {
  (void)in_sizes; (void)n_in; (void)out_size; (void)ws_size;
  const int*   Z      = (const int*)d_in[0];
  const int*   nbr    = (const int*)d_in[1];
  const float* disp   = (const float*)d_in[2];
  const float* emb    = (const float*)d_in[3];
  const float* Wembt  = (const float*)d_in[4];
  const float* Wrad   = (const float*)d_in[5];
  const float* W_sets[6] = { (const float*)d_in[6],  (const float*)d_in[7],
                             (const float*)d_in[9],  (const float*)d_in[10],
                             (const float*)d_in[11], (const float*)d_in[13] };
  const float* wp0 = (const float*)d_in[8];
  const float* wp1 = (const float*)d_in[12];
  const float* twt = (const float*)d_in[14];
  float* out = (float*)d_out;

  char* ws = (char*)d_ws;
  int*      nnz     = (int*)(ws + OFF_NNZ);
  CGEntry*  entries = (CGEntry*)(ws + OFF_ENTRIES);
  __bf16*   Wt      = (__bf16*)(ws + OFF_WT);       // 6 sets, 40960 B each
  float*    Aatom   = (float*)(ws + OFF_AATOM);
  __bf16*   bufY    = (__bf16*)(ws + OFF_BUF);
  __bf16*   bufU    = (__bf16*)(ws + OFF_BUF + BUF_BYTES);
  __bf16*   bufV    = (__bf16*)(ws + OFF_BUF + 2 * BUF_BYTES);

  const int tiles = NUM_LM * NUM_EDGES / 32;        // 31250 (32-row tiles)
  const int ddBlk = (tiles + 7) / 8;                // 8 waves / block

  // tables + weight conversion
  init_cg_tables<<<1, 32, 0, stream>>>(entries, nnz);
  for (int s = 0; s < 6; ++s)
    wconv_kernel<<<(5 * 64 * 64 + 255) / 256, 256, 0, stream>>>(W_sets[s], Wt + (size_t)s * 5 * 4096);

  // radial features + atom embedding + output init
  radial_kernel<<<NUM_EDGES / 4, 256, 0, stream>>>(disp, nbr, Z, Wrad, bufY);
  atom_embed<<<(NUM_ATOMS * FEAT + 255) / 256, 256, 0, stream>>>(Z, emb, Wembt, Aatom);
  out_init<<<(NUM_ATOMS * NUM_LM * FEAT + 255) / 256, 256, 0, stream>>>(Aatom, out);

  // layer 0: Y -> U, V ; cg -> Y ; Wout -> U
  deg_dense_wmma<<<ddBlk, 256, 0, stream>>>(bufY, Wt + 0 * 5 * 4096, bufU, tiles);
  deg_dense_wmma<<<ddBlk, 256, 0, stream>>>(bufY, Wt + 1 * 5 * 4096, bufV, tiles);
  cg_kernel<<<NUM_EDGES / 2, 128, 0, stream>>>(bufU, bufV, wp0, bufY, entries, nnz);
  deg_dense_wmma<<<ddBlk, 256, 0, stream>>>(bufY, Wt + 2 * 5 * 4096, bufU, tiles);

  // layer 1: U -> Y, V ; cg -> U ; Wout -> Y
  deg_dense_wmma<<<ddBlk, 256, 0, stream>>>(bufU, Wt + 3 * 5 * 4096, bufY, tiles);
  deg_dense_wmma<<<ddBlk, 256, 0, stream>>>(bufU, Wt + 4 * 5 * 4096, bufV, tiles);
  cg_kernel<<<NUM_EDGES / 2, 128, 0, stream>>>(bufY, bufV, wp1, bufU, entries, nnz);
  deg_dense_wmma<<<ddBlk, 256, 0, stream>>>(bufU, Wt + 5 * 5 * 4096, bufY, tiles);

  // scale + segment-sum
  scatter_kernel<<<NUM_EDGES / 4, 256, 0, stream>>>(bufY, Aatom, twt, nbr, out);
}
